// LinearWrapper_30451318128976
// MI455X (gfx1250) — matrix operand order repaired and checked by reading, not compiled, not run on hardware
//
#include <hip/hip_runtime.h>

// out = x @ (W + s * B@A)^T + bias
//     = x @ W^T  +  (s * x @ A^T) @ B^T  +  bias
// M=8192 (tokens), N=4096 (d_out), K=4096 (d_in), R=16
//
// Primary path (needs ~192.5 MB workspace):
//   1) t = s * x @ A^T                         (f32, [8192,16])
//   2) split x, W once into bf16 hi/lo planes  (RNE hi + bf16(residual) lo)
//   3) main GEMM: global -> LDS via CDNA5 async loads (ASYNCcnt), double-buffered,
//      3x v_wmma_f32_16x16x32_bf16 per tile (hi*hi + hi*lo + lo*hi),
//      LoRA epilogue via exact v_wmma_f32_16x16x4_f32 over R=16.
// Fallback path (small workspace): round-1 kernel with in-loop conversion.

#define M_TOK 8192
#define N_OUT 4096
#define K_IN  4096
#define R_LORA 16

#define BM 128
#define BN 64
#define BK 32
#define LDSP 40          // padded row stride in bf16 elements (80 bytes, conflict-free)

typedef __attribute__((ext_vector_type(16))) __bf16 v16bf;
typedef __attribute__((ext_vector_type(8)))  float  v8f;
typedef __attribute__((ext_vector_type(2)))  float  v2f;

// ---------------- conversion helpers ----------------

__device__ __forceinline__ unsigned rne_bf16(float f) {
    unsigned u = __float_as_uint(f);
    return (u + 0x7FFFu + ((u >> 16) & 1u)) >> 16;   // round-to-nearest-even
}

// float4 -> packed bf16 hi pair + residual-lo pair
__device__ __forceinline__ void cvt4(const float4 f, uint2& h, uint2& l) {
    unsigned h0 = rne_bf16(f.x), h1 = rne_bf16(f.y);
    unsigned h2 = rne_bf16(f.z), h3 = rne_bf16(f.w);
    float r0 = f.x - __uint_as_float(h0 << 16);
    float r1 = f.y - __uint_as_float(h1 << 16);
    float r2 = f.z - __uint_as_float(h2 << 16);
    float r3 = f.w - __uint_as_float(h3 << 16);
    h = make_uint2(h0 | (h1 << 16), h2 | (h3 << 16));
    l = make_uint2(rne_bf16(r0) | (rne_bf16(r1) << 16),
                   rne_bf16(r2) | (rne_bf16(r3) << 16));
}

// ---------------- fragment loaders (ISA 7.12.2 layouts) ----------------

// A fragment (16x32 bf16): lanes 0-15 take K0..7 & K16..23 -> byte offs {h*16, h*16+32}
__device__ __forceinline__ v16bf ld_fragA(const unsigned short* base, int row, int half) {
    union { uint4 q[2]; v16bf v; } u;
    const char* p = (const char*)base + row * (LDSP * 2) + half * 16;
    u.q[0] = *(const uint4*)(p);
    u.q[1] = *(const uint4*)(p + 32);
    return u.v;
}

// B fragment (32x16 bf16): lanes 0-15 take K0..7 & K16..23 -> byte offs {h*16, h*16+32}
__device__ __forceinline__ v16bf ld_fragB(const unsigned short* base, int row, int half) {
    union { uint4 q[2]; v16bf v; } u;
    const char* p = (const char*)base + row * (LDSP * 2) + half * 16;
    u.q[0] = *(const uint4*)(p);
    u.q[1] = *(const uint4*)(p + 32);
    return u.v;
}

// ---------------- CDNA5 async global->LDS copy (ASYNCcnt) ----------------

__device__ __forceinline__ void async_b128(const unsigned short* lds_ptr,
                                           const unsigned short* g_ptr) {
    // Low 32 bits of the generic pointer to a __shared__ object are the LDS byte
    // offset (flat shared aperture: addr[31:0] = LDS offset).
    unsigned lds_off = (unsigned)(unsigned long long)lds_ptr;
    asm volatile("global_load_async_to_lds_b128 %0, %1, off"
                 :: "v"(lds_off), "v"(g_ptr) : "memory");
}

__device__ __forceinline__ void wait_async0() {
    asm volatile("s_wait_asynccnt 0" ::: "memory");
}

// ---------------- fallback-path tile staging (register round-trip) ----------------

__device__ __forceinline__ void ld_tiles(const float* xg, const float* wg,
                                         int row_t, int k0,
                                         float4* xr, float4* wr) {
#pragma unroll
    for (int i = 0; i < 4; ++i)
        xr[i] = *(const float4*)(xg + (size_t)(row_t + 32 * i) * K_IN + k0);
#pragma unroll
    for (int i = 0; i < 2; ++i)
        wr[i] = *(const float4*)(wg + (size_t)(row_t + 32 * i) * K_IN + k0);
}

__device__ __forceinline__ void stage_store(unsigned short* xhi, unsigned short* xlo,
                                            unsigned short* whi, unsigned short* wlo,
                                            int row_t, int c4,
                                            const float4* xr, const float4* wr) {
#pragma unroll
    for (int i = 0; i < 4; ++i) {
        uint2 h, l; cvt4(xr[i], h, l);
        int r = row_t + 32 * i;
        *(uint2*)((char*)xhi + r * (LDSP * 2) + c4 * 8) = h;
        *(uint2*)((char*)xlo + r * (LDSP * 2) + c4 * 8) = l;
    }
#pragma unroll
    for (int i = 0; i < 2; ++i) {
        uint2 h, l; cvt4(wr[i], h, l);
        int r = row_t + 32 * i;
        *(uint2*)((char*)whi + r * (LDSP * 2) + c4 * 8) = h;
        *(uint2*)((char*)wlo + r * (LDSP * 2) + c4 * 8) = l;
    }
}

// ---------------- kernel: t = scale * x @ A^T  ([M_TOK, 16] f32) ----------------

__global__ __launch_bounds__(256)
void lora_xA_kernel(const float* __restrict__ x, const float* __restrict__ A,
                    const float* __restrict__ scale_p, float* __restrict__ t) {
    int tid = blockIdx.x * blockDim.x + threadIdx.x;   // 0 .. M_TOK*16-1
    int r   = tid & (R_LORA - 1);
    int row = tid >> 4;
    const float4* xr = (const float4*)(x + (size_t)row * K_IN);
    const float4* ar = (const float4*)(A + (size_t)r * K_IN);
    float s = 0.f;
#pragma unroll 8
    for (int k = 0; k < K_IN / 4; ++k) {
        float4 xv = xr[k], av = ar[k];
        s = fmaf(xv.x, av.x, s); s = fmaf(xv.y, av.y, s);
        s = fmaf(xv.z, av.z, s); s = fmaf(xv.w, av.w, s);
    }
    t[(size_t)row * R_LORA + r] = s * scale_p[0];
}

// ---------------- kernel: split f32 -> bf16 hi/lo planes ----------------

__global__ __launch_bounds__(256)
void cvt_split_kernel(const float* __restrict__ src,
                      unsigned short* __restrict__ hi,
                      unsigned short* __restrict__ lo) {
    size_t i = (size_t)blockIdx.x * blockDim.x + threadIdx.x;   // one float4 each
    float4 f = ((const float4*)src)[i];
    uint2 h, l; cvt4(f, h, l);
    ((uint2*)hi)[i] = h;
    ((uint2*)lo)[i] = l;
}

// ---------------- shared epilogue: += t @ Bl^T (f32 WMMA), += bias, store ----------

__device__ __forceinline__ void epilogue_tile(v8f c, int tm, int tn,
                                              int lrow, int half,
                                              const float* __restrict__ t,
                                              const float* __restrict__ Bl,
                                              const float* __restrict__ bias,
                                              float* __restrict__ out) {
    const float* trow = t  + (size_t)(tm + lrow) * R_LORA;   // A operand: 16x4 f32
    const float* brow = Bl + (size_t)(tn + lrow) * R_LORA;   // B operand: 4x16 f32
#pragma unroll
    for (int s4 = 0; s4 < 4; ++s4) {
        int ka = s4 * 4 + half * 2;
        v2f av; av[0] = trow[ka]; av[1] = trow[ka + 1];
        v2f bv; bv[0] = brow[ka]; bv[1] = brow[ka + 1];
        c = __builtin_amdgcn_wmma_f32_16x16x4_f32(
            false, av, false, bv, (short)0, c, false, false);
    }
    const float bval = bias[tn + lrow];
#pragma unroll
    for (int j = 0; j < 8; ++j) {
        int m = tm + j + 8 * half;
        out[(size_t)m * N_OUT + tn + lrow] = c[j] + bval;
    }
}

// ---------------- primary kernel: async-fed split-bf16 WMMA GEMM ----------------

__global__ __launch_bounds__(256)
void lora_gemm_async_kernel(const unsigned short* __restrict__ xh,
                            const unsigned short* __restrict__ xl,
                            const unsigned short* __restrict__ wh,
                            const unsigned short* __restrict__ wl,
                            const float* __restrict__ bias,
                            const float* __restrict__ Bl,
                            const float* __restrict__ t,
                            float* __restrict__ out) {
    __shared__ __align__(16) unsigned short xs [2][2][BM * LDSP];  // [buf][hi/lo]
    __shared__ __align__(16) unsigned short wls[2][2][BN * LDSP];

    const int tid  = threadIdx.x;
    const int lane = tid & 31;
    const int wave = tid >> 5;
    const int wm   = wave & 3;        // 4 waves along M
    const int wn   = wave >> 2;       // 2 waves along N
    const int lrow = lane & 15;
    const int half = lane >> 4;

    const int bm = blockIdx.y * BM;
    const int bn = blockIdx.x * BN;

    // async-copy mapping: each thread moves 6 x 16B per stage
    const int seg = tid & 3;          // 16B segment within a 32-elem (64B) row
    const int rr  = tid >> 2;         // 0..63

    const unsigned short* xh_g = xh + (size_t)(bm + rr) * K_IN + seg * 8;
    const unsigned short* xl_g = xl + (size_t)(bm + rr) * K_IN + seg * 8;
    const unsigned short* wh_g = wh + (size_t)(bn + rr) * K_IN + seg * 8;
    const unsigned short* wl_g = wl + (size_t)(bn + rr) * K_IN + seg * 8;
    const size_t rowskip = (size_t)64 * K_IN;

    const int l0 = rr * LDSP + seg * 8;
    const int l1 = (rr + 64) * LDSP + seg * 8;

#define ISSUE_STAGE(buf, k0)                                    \
    do {                                                        \
        async_b128(&xs [buf][0][l0], xh_g + (k0));              \
        async_b128(&xs [buf][0][l1], xh_g + rowskip + (k0));    \
        async_b128(&xs [buf][1][l0], xl_g + (k0));              \
        async_b128(&xs [buf][1][l1], xl_g + rowskip + (k0));    \
        async_b128(&wls[buf][0][l0], wh_g + (k0));              \
        async_b128(&wls[buf][1][l0], wl_g + (k0));              \
    } while (0)

    ISSUE_STAGE(0, 0);
    wait_async0();
    __syncthreads();

    v8f acc[2][2] = {};

    const int NSTEP = K_IN / BK;      // 128
    for (int s = 0; s < NSTEP; ++s) {
        const int cur = s & 1;
        if (s < NSTEP - 1) ISSUE_STAGE(cur ^ 1, (s + 1) * BK);

        v16bf ah[2], al[2], bh[2], bl[2];
#pragma unroll
        for (int mi = 0; mi < 2; ++mi) {
            int r = wm * 32 + mi * 16 + lrow;
            ah[mi] = ld_fragA(xs[cur][0], r, half);
            al[mi] = ld_fragA(xs[cur][1], r, half);
        }
#pragma unroll
        for (int ni = 0; ni < 2; ++ni) {
            int r = wn * 32 + ni * 16 + lrow;
            bh[ni] = ld_fragB(wls[cur][0], r, half);
            bl[ni] = ld_fragB(wls[cur][1], r, half);
        }
#pragma unroll
        for (int mi = 0; mi < 2; ++mi)
#pragma unroll
            for (int ni = 0; ni < 2; ++ni) {
                acc[mi][ni] = __builtin_amdgcn_wmma_f32_16x16x32_bf16(
                    false, ah[mi], false, bh[ni], (short)0, acc[mi][ni], false, false);
                acc[mi][ni] = __builtin_amdgcn_wmma_f32_16x16x32_bf16(
                    false, ah[mi], false, bl[ni], (short)0, acc[mi][ni], false, false);
                acc[mi][ni] = __builtin_amdgcn_wmma_f32_16x16x32_bf16(
                    false, al[mi], false, bh[ni], (short)0, acc[mi][ni], false, false);
            }

        if (s < NSTEP - 1) wait_async0();   // next stage fully resident in LDS
        __syncthreads();
    }
#undef ISSUE_STAGE

#pragma unroll
    for (int mi = 0; mi < 2; ++mi)
#pragma unroll
        for (int ni = 0; ni < 2; ++ni)
            epilogue_tile(acc[mi][ni],
                          bm + wm * 32 + mi * 16, bn + wn * 32 + ni * 16,
                          lrow, half, t, Bl, bias, out);
}

// ---------------- fallback kernel: in-loop conversion (round-1 path) ----------------

__global__ __launch_bounds__(256)
void lora_gemm_kernel(const float* __restrict__ x, const float* __restrict__ W,
                      const float* __restrict__ bias, const float* __restrict__ Bl,
                      const float* __restrict__ t, float* __restrict__ out) {
    __shared__ __align__(16) unsigned short xs [2][2][BM * LDSP];
    __shared__ __align__(16) unsigned short wls[2][2][BN * LDSP];

    const int tid  = threadIdx.x;
    const int lane = tid & 31;
    const int wave = tid >> 5;
    const int wm   = wave & 3;
    const int wn   = wave >> 2;
    const int lrow = lane & 15;
    const int half = lane >> 4;

    const int bm = blockIdx.y * BM;
    const int bn = blockIdx.x * BN;

    const int row_t = tid >> 3;
    const int c4    = tid & 7;

    const float* xg = x + (size_t)bm * K_IN + c4 * 4;
    const float* wg = W + (size_t)bn * K_IN + c4 * 4;

    float4 xr[4], wr[2];
    ld_tiles(xg, wg, row_t, 0, xr, wr);
    stage_store(xs[0][0], xs[0][1], wls[0][0], wls[0][1], row_t, c4, xr, wr);
    __syncthreads();

    v8f acc[2][2] = {};

    const int NSTEP = K_IN / BK;
    for (int s = 0; s < NSTEP; ++s) {
        const int cur = s & 1;
        if (s < NSTEP - 1) ld_tiles(xg, wg, row_t, (s + 1) * BK, xr, wr);

        v16bf ah[2], al[2], bh[2], bl[2];
#pragma unroll
        for (int mi = 0; mi < 2; ++mi) {
            int r = wm * 32 + mi * 16 + lrow;
            ah[mi] = ld_fragA(xs[cur][0], r, half);
            al[mi] = ld_fragA(xs[cur][1], r, half);
        }
#pragma unroll
        for (int ni = 0; ni < 2; ++ni) {
            int r = wn * 32 + ni * 16 + lrow;
            bh[ni] = ld_fragB(wls[cur][0], r, half);
            bl[ni] = ld_fragB(wls[cur][1], r, half);
        }
#pragma unroll
        for (int mi = 0; mi < 2; ++mi)
#pragma unroll
            for (int ni = 0; ni < 2; ++ni) {
                acc[mi][ni] = __builtin_amdgcn_wmma_f32_16x16x32_bf16(
                    false, ah[mi], false, bh[ni], (short)0, acc[mi][ni], false, false);
                acc[mi][ni] = __builtin_amdgcn_wmma_f32_16x16x32_bf16(
                    false, ah[mi], false, bl[ni], (short)0, acc[mi][ni], false, false);
                acc[mi][ni] = __builtin_amdgcn_wmma_f32_16x16x32_bf16(
                    false, al[mi], false, bh[ni], (short)0, acc[mi][ni], false, false);
            }

        if (s < NSTEP - 1)
            stage_store(xs[cur ^ 1][0], xs[cur ^ 1][1],
                        wls[cur ^ 1][0], wls[cur ^ 1][1], row_t, c4, xr, wr);
        __syncthreads();
    }

#pragma unroll
    for (int mi = 0; mi < 2; ++mi)
#pragma unroll
        for (int ni = 0; ni < 2; ++ni)
            epilogue_tile(acc[mi][ni],
                          bm + wm * 32 + mi * 16, bn + wn * 32 + ni * 16,
                          lrow, half, t, Bl, bias, out);
}

// ---------------- launcher ----------------

extern "C" void kernel_launch(void* const* d_in, const int* in_sizes, int n_in,
                              void* d_out, int out_size, void* d_ws, size_t ws_size,
                              hipStream_t stream) {
    const float* x     = (const float*)d_in[0];   // [8192, 4096]
    const float* W     = (const float*)d_in[1];   // [4096, 4096]
    const float* bias  = (const float*)d_in[2];   // [4096]
    const float* lB    = (const float*)d_in[3];   // [4096, 16]
    const float* lA    = (const float*)d_in[4];   // [16, 4096]
    const float* scale = (const float*)d_in[5];   // [1]
    float*       out   = (float*)d_out;           // [8192, 4096]

    const size_t t_bytes  = (size_t)M_TOK * R_LORA * sizeof(float);        // 512 KB
    const size_t x_plane  = (size_t)M_TOK * K_IN * sizeof(unsigned short); // 64 MB
    const size_t w_plane  = (size_t)N_OUT * K_IN * sizeof(unsigned short); // 32 MB
    const size_t need     = t_bytes + 2 * x_plane + 2 * w_plane;           // ~192.5 MB

    float* t = (float*)d_ws;

    // t = scale * x @ A^T
    lora_xA_kernel<<<(M_TOK * R_LORA) / 256, 256, 0, stream>>>(x, lA, scale, t);

    dim3 grid(N_OUT / BN, M_TOK / BM);

    if (ws_size >= need) {
        char* p = (char*)d_ws + t_bytes;
        unsigned short* xh = (unsigned short*)p;               p += x_plane;
        unsigned short* xl = (unsigned short*)p;               p += x_plane;
        unsigned short* wh = (unsigned short*)p;               p += w_plane;
        unsigned short* wl = (unsigned short*)p;

        cvt_split_kernel<<<(M_TOK * K_IN / 4) / 256, 256, 0, stream>>>(x, xh, xl);
        cvt_split_kernel<<<(N_OUT * K_IN / 4) / 256, 256, 0, stream>>>(W, wh, wl);

        lora_gemm_async_kernel<<<grid, 256, 0, stream>>>(xh, xl, wh, wl,
                                                         bias, lB, t, out);
    } else {
        lora_gemm_kernel<<<grid, 256, 0, stream>>>(x, W, bias, lB, t, out);
    }
}
